// SparseFrameAttentionModule_72275709657158
// MI455X (gfx1250) — compile-verified, matrix operand order checked by reading, not run on hardware
//
#include <hip/hip_runtime.h>

// Sparse frame attention for MI455X (gfx1250, wave32, WMMA f16 16x16x32).
// B=1, F=40 frames, HW=390 tokens/frame, H=8 heads, HD=64, SEL=4 (self+top3).

#define F_N    40
#define HW_N   390
#define H_N    8
#define DIM_N  512
#define HD_N   64
#define TOPK_N 3
#define SEL_N  4
#define SKV_N  (SEL_N * HW_N)              // 1560 keys per query frame
#define CH_N   64                          // keys per KV chunk
#define NCH_N  ((SKV_N + CH_N - 1) / CH_N) // 25 chunks
#define NW_N   25                          // waves per block = query tiles (25*16 >= 390)
#define NTHR   (NW_N * 32)                 // 800 threads

typedef _Float16 half_t;
typedef __attribute__((ext_vector_type(16))) _Float16 v16h;
typedef __attribute__((ext_vector_type(8)))  _Float16 v8h;
typedef __attribute__((ext_vector_type(4)))  _Float16 v4h;
typedef __attribute__((ext_vector_type(8)))  float    v8f;

// Load one WMMA A/B fragment (16x32 f16, mirrored per-lane layout) from
// row-major LDS. rowptr = start of this lane's 32-element contraction row.
// Lane layout: halfs 0..7 = k = hi*8..+7 ; halfs 8..15 = k = 16+hi*8.. (hi = lane>>4).
__device__ __forceinline__ v16h ld_frag_lds(const half_t* rowptr, int hi) {
  v8h a0 = *(const v8h*)(rowptr + hi * 8);
  v8h a1 = *(const v8h*)(rowptr + 16 + hi * 8);
  v16h r;
#pragma unroll
  for (int j = 0; j < 8; ++j) { r[j] = a0[j]; r[j + 8] = a1[j]; }
  return r;
}

// Same fragment pattern, loaded directly from global f32 with scale + f16 convert.
__device__ __forceinline__ v16h ld_frag_gf32(const float* rowptr, int hi, float scale) {
  v16h r;
#pragma unroll
  for (int j = 0; j < 8; ++j) {
    r[j]     = (half_t)(rowptr[hi * 8 + j] * scale);
    r[j + 8] = (half_t)(rowptr[16 + hi * 8 + j] * scale);
  }
  return r;
}

// --- Kernel 1: top-k frame selection (tiny) --------------------------------
__global__ __launch_bounds__(64) void topk_sel_kernel(const float* __restrict__ sim,
                                                      int* __restrict__ sel) {
  int fr = threadIdx.x;
  if (fr >= F_N) return;
  unsigned long long used = 1ull << fr;   // mask self (reference adds -1e6 on diag)
  sel[fr * SEL_N + 0] = fr;               // self frame first
#pragma unroll
  for (int kk = 0; kk < TOPK_N; ++kk) {
    float best = -3.4e38f; int bi = 0;
    for (int j = 0; j < F_N; ++j) {
      if (!((used >> j) & 1ull)) {
        float vv = sim[fr * F_N + j];
        if (vv > best) { best = vv; bi = j; }   // strict > keeps lowest index on ties
      }
    }
    used |= 1ull << bi;
    sel[fr * SEL_N + 1 + kk] = bi;
  }
}

// --- Kernel 2: flash attention over gathered frames ------------------------
__global__ __launch_bounds__(NTHR) void sparse_frame_attn_kernel(
    const float* __restrict__ qg, const float* __restrict__ kg,
    const float* __restrict__ vg, const int* __restrict__ sel,
    float* __restrict__ outg) {
  // Padded LDS tiles: strides are multiples of 8 halfs (16B alignment for b128
  // fragment reads) and spread 16 consecutive rows across banks.
  __shared__ __align__(16) half_t Ks[CH_N][72];       // K chunk, [key][d], 64 keys
  __shared__ __align__(16) half_t Vt[HD_N][72];       // V chunk transposed, [d][key]
  __shared__ __align__(16) half_t Ps[NW_N][16][40];   // per-wave P tile (32 keys, reused 2x)
  __shared__ int sel_s[SEL_N];

  const int f    = blockIdx.x / H_N;
  const int h    = blockIdx.x % H_N;
  const int tid  = threadIdx.x;
  const int w    = tid >> 5;     // wave id = query tile id
  const int lane = tid & 31;
  const int ln   = lane & 15;
  const int hf   = lane >> 4;

  if (tid < SEL_N) sel_s[tid] = sel[f * SEL_N + tid];

  // Persistent Q fragments for this wave's 16-query tile (scale = 1/sqrt(64)).
  const int q0 = w * 16;
  int qr = q0 + ln; if (qr >= HW_N) qr = HW_N - 1;     // clamp tail rows (not stored)
  const float* qrow = qg + ((size_t)(f * HW_N + qr)) * DIM_N + h * HD_N;
  const v16h qf0 = ld_frag_gf32(qrow,      hf, 0.125f);   // d = 0..31
  const v16h qf1 = ld_frag_gf32(qrow + 32, hf, 0.125f);   // d = 32..63

  // Online-softmax state. Slot r <-> query m = r + 8*hf (replicated over the
  // 16 lanes of each half-wave, matching the f32 C/D layout).
  float m_[8], l_[8];
  v8f acc[4];
#pragma unroll
  for (int r = 0; r < 8; ++r) { m_[r] = -1e30f; l_[r] = 0.f; }
#pragma unroll
  for (int dt = 0; dt < 4; ++dt)
#pragma unroll
    for (int r = 0; r < 8; ++r) acc[dt][r] = 0.f;

  for (int c = 0; c < NCH_N; ++c) {
    const int kbase = c * CH_N;
    __syncthreads();                       // previous chunk fully consumed
    // Cooperative gather: 64 keys x 64 dims of K and V, float4 per thread,
    // one frame-gather division per thread. Prefetch next chunk's lines.
    for (int i = tid; i < CH_N * (HD_N / 4); i += NTHR) {   // 1024 float4 loads
      int kk = i >> 4, d4 = (i & 15) << 2;
      int g = kbase + kk; if (g >= SKV_N) g = SKV_N - 1;    // clamped, masked later
      int s = g / HW_N; int off = g - s * HW_N;
      size_t base = ((size_t)(sel_s[s] * HW_N + off)) * DIM_N + h * HD_N + d4;
      float4 kx = *(const float4*)(kg + base);
      float4 vx = *(const float4*)(vg + base);
      v4h kh; kh[0] = (half_t)kx.x; kh[1] = (half_t)kx.y;
              kh[2] = (half_t)kx.z; kh[3] = (half_t)kx.w;
      *(v4h*)&Ks[kk][d4] = kh;
      Vt[d4 + 0][kk] = (half_t)vx.x;
      Vt[d4 + 1][kk] = (half_t)vx.y;
      Vt[d4 + 2][kk] = (half_t)vx.z;
      Vt[d4 + 3][kk] = (half_t)vx.w;
      int g2 = g + CH_N;                   // prefetch next chunk (L2-resident stream)
      if (g2 < SKV_N) {
        int s2 = g2 / HW_N; int off2 = g2 - s2 * HW_N;
        size_t base2 = ((size_t)(sel_s[s2] * HW_N + off2)) * DIM_N + h * HD_N + d4;
        __builtin_prefetch(kg + base2, 0, 1);
        __builtin_prefetch(vg + base2, 0, 1);
      }
    }
    __syncthreads();                       // chunk ready

    // --- scores S = (Q/8) K^T for four 16-key subtiles --------------------
    v8f sc[4];
#pragma unroll
    for (int j = 0; j < 4; ++j) {
#pragma unroll
      for (int r = 0; r < 8; ++r) sc[j][r] = 0.f;
      v16h kf0 = ld_frag_lds(&Ks[j * 16 + ln][0], hf);
      sc[j] = __builtin_amdgcn_wmma_f32_16x16x32_f16(false, qf0, false, kf0,
                                                     (short)0, sc[j], false, false);
      v16h kf1 = ld_frag_lds(&Ks[j * 16 + ln][32], hf);
      sc[j] = __builtin_amdgcn_wmma_f32_16x16x32_f16(false, qf1, false, kf1,
                                                     (short)0, sc[j], false, false);
      if (kbase + j * 16 + ln >= SKV_N) {  // mask invalid tail keys (lane = key col)
#pragma unroll
        for (int r = 0; r < 8; ++r) sc[j][r] = -1e30f;
      }
    }

    // --- online softmax over 64 keys (row reduce across 16 key-lanes) -----
    float af[8], p2s[8], p3s[8];
#pragma unroll
    for (int r = 0; r < 8; ++r) {
      float tmax = fmaxf(fmaxf(sc[0][r], sc[1][r]), fmaxf(sc[2][r], sc[3][r]));
#pragma unroll
      for (int o = 1; o < 16; o <<= 1) tmax = fmaxf(tmax, __shfl_xor(tmax, o, 32));
      float mnew  = fmaxf(m_[r], tmax);
      float alpha = __expf(m_[r] - mnew);
      float p0 = __expf(sc[0][r] - mnew);
      float p1 = __expf(sc[1][r] - mnew);
      float p2 = __expf(sc[2][r] - mnew);
      float p3 = __expf(sc[3][r] - mnew);
      float rs = (p0 + p1) + (p2 + p3);
#pragma unroll
      for (int o = 1; o < 16; o <<= 1) rs += __shfl_xor(rs, o, 32);
      l_[r] = alpha * l_[r] + rs;
      m_[r] = mnew;
      af[r] = alpha;
      int qm = r + 8 * hf;                 // C-layout -> A-layout via LDS
      Ps[w][qm][ln]      = (half_t)p0;     // keys 0..15
      Ps[w][qm][16 + ln] = (half_t)p1;     // keys 16..31
      p2s[r] = p2; p3s[r] = p3;            // keys 32..63 staged after first PV
    }
    asm volatile("" ::: "memory");         // keep LDS store->load order (same wave,
                                           // hardware LDS is in-order per wave)

    // rescale running accumulators by exp(m_old - m_new), once per 64 keys
#pragma unroll
    for (int dt = 0; dt < 4; ++dt)
#pragma unroll
      for (int r = 0; r < 8; ++r) acc[dt][r] *= af[r];

    // --- O += P V, keys 0..31 ---------------------------------------------
    {
      v16h pf = ld_frag_lds(&Ps[w][ln][0], hf);
#pragma unroll
      for (int dt = 0; dt < 4; ++dt) {
        v16h vf = ld_frag_lds(&Vt[dt * 16 + ln][0], hf);
        acc[dt] = __builtin_amdgcn_wmma_f32_16x16x32_f16(false, pf, false, vf,
                                                         (short)0, acc[dt], false, false);
      }
    }
    asm volatile("" ::: "memory");

    // stage P for keys 32..63 into the same per-wave buffer (in-order LDS)
#pragma unroll
    for (int r = 0; r < 8; ++r) {
      int qm = r + 8 * hf;
      Ps[w][qm][ln]      = (half_t)p2s[r];
      Ps[w][qm][16 + ln] = (half_t)p3s[r];
    }
    asm volatile("" ::: "memory");

    // --- O += P V, keys 32..63 --------------------------------------------
    {
      v16h pf = ld_frag_lds(&Ps[w][ln][0], hf);
#pragma unroll
      for (int dt = 0; dt < 4; ++dt) {
        v16h vf = ld_frag_lds(&Vt[dt * 16 + ln][32], hf);
        acc[dt] = __builtin_amdgcn_wmma_f32_16x16x32_f16(false, pf, false, vf,
                                                         (short)0, acc[dt], false, false);
      }
    }
  }

  // --- epilogue: out = acc / l ---------------------------------------------
#pragma unroll
  for (int r = 0; r < 8; ++r) {
    int ql = q0 + r + 8 * hf;
    if (ql < HW_N) {
      float inv = 1.f / l_[r];
      size_t orow = ((size_t)(f * HW_N + ql)) * DIM_N + h * HD_N;
#pragma unroll
      for (int dt = 0; dt < 4; ++dt)
        outg[orow + dt * 16 + ln] = acc[dt][r] * inv;   // lane = d column
    }
  }
}

extern "C" void kernel_launch(void* const* d_in, const int* in_sizes, int n_in,
                              void* d_out, int out_size, void* d_ws, size_t ws_size,
                              hipStream_t stream) {
  const float* q   = (const float*)d_in[0];
  const float* k   = (const float*)d_in[1];
  const float* v   = (const float*)d_in[2];
  const float* sim = (const float*)d_in[3];
  float* out = (float*)d_out;
  int*   sel = (int*)d_ws;                 // F_N * SEL_N ints

  topk_sel_kernel<<<1, 64, 0, stream>>>(sim, sel);
  sparse_frame_attn_kernel<<<dim3(F_N * H_N), dim3(NTHR), 0, stream>>>(q, k, v, sel, out);
}
